// MultiHeadAttention_53034256171685
// MI455X (gfx1250) — matrix operand order repaired and checked by reading, not compiled, not run on hardware
//
#include <hip/hip_runtime.h>
#include <hip/hip_bf16.h>

#define DEV __device__ __forceinline__

typedef __attribute__((ext_vector_type(16))) __bf16 v16bf;
typedef __attribute__((ext_vector_type(8)))  __bf16 v8bf;
typedef __attribute__((ext_vector_type(8)))  float  v8f;
typedef __attribute__((ext_vector_type(4)))  float  v4f;

constexpr int Bb = 4, Tt = 2048, Cc = 768, Hh = 12, Dd = 64;
constexpr int Mrows = Bb * Tt; // 8192

DEV v8f wmma_bf16(v16bf a, v16bf b, v8f c) {
  return __builtin_amdgcn_wmma_f32_16x16x32_bf16(false, a, false, b, (short)0, c, false, false);
}

DEV v8f zero_v8f() {
  v8f z;
#pragma unroll
  for (int i = 0; i < 8; ++i) z[i] = 0.0f;
  return z;
}

// Build a 16-element bf16 fragment from two 16-byte aligned runs.
DEV v16bf ld_frag(const __bf16* p0, const __bf16* p1) {
  v8bf lo = *(const v8bf*)p0;
  v8bf hi = *(const v8bf*)p1;
  v16bf r;
#pragma unroll
  for (int i = 0; i < 8; ++i) { r[i] = lo[i]; r[i + 8] = hi[i]; }
  return r;
}

// Wave-relative LDS byte offset of a generic shared pointer: generic shared
// addresses are {SHARED_BASE aperture in [63:32] | lds_offset in [31:0]}
// (ISA 10.2 per-aperture calc), so truncation yields the DS address.
DEV unsigned lds_off(const void* p) { return (unsigned)(unsigned long long)(uintptr_t)p; }

// Async global->LDS copy of 32 bytes (two B128 ops), tracked on ASYNCcnt.
DEV void async_copy_b256(unsigned lds_byte_off, const void* gsrc) {
  unsigned long long ga = (unsigned long long)(uintptr_t)gsrc;
  asm volatile("global_load_async_to_lds_b128 %0, %1, off\n\t"
               "global_load_async_to_lds_b128 %0, %1, off offset:16"
               :: "v"(lds_byte_off), "v"(ga) : "memory");
}

DEV void async_copy_b512(unsigned lds_byte_off, const void* gsrc) {
  unsigned long long ga = (unsigned long long)(uintptr_t)gsrc;
  asm volatile("global_load_async_to_lds_b128 %0, %1, off\n\t"
               "global_load_async_to_lds_b128 %0, %1, off offset:16\n\t"
               "global_load_async_to_lds_b128 %0, %1, off offset:32\n\t"
               "global_load_async_to_lds_b128 %0, %1, off offset:48"
               :: "v"(lds_byte_off), "v"(ga) : "memory");
}

DEV void wait_async0() { asm volatile("s_wait_asynccnt 0x0" ::: "memory"); }

// ---------------------------------------------------------------------------
// GEMM: Out = A[M,K] @ W[N,K]^T   (K = N = Cc = 768, M = 8192), K-step 64.
//   A_BF16:  A is bf16 (workspace Y, staged via async-to-LDS) vs fp32 (inputs,
//            converted to bf16 while staging).
//   SCATTER: epilogue writes bf16 into [B,H,T,D] head layout (projections)
//            else fp32 row-major [M,N] (final output)
// ---------------------------------------------------------------------------
template <bool A_BF16, bool SCATTER>
__global__ __launch_bounds__(256)
void gemm_xwT_kernel(const void* __restrict__ Ap,
                     const float* __restrict__ W,
                     void* __restrict__ Op) {
  __shared__ __align__(16) __bf16 sA[128][72];   // [m][k],  stride 144B
  __shared__ __align__(16) __bf16 sB[64][136];   // [k][n] = W^T tile, stride 272B

  const int tid  = threadIdx.x;
  const int lane = tid & 31;
  const int wv   = tid >> 5;
  const int half = lane >> 4;
  const int ln15 = lane & 15;
  const int wm   = wv & 3;        // 4 waves along M
  const int wn   = wv >> 2;       // 2 waves along N
  const int m0   = blockIdx.x * 128;
  const int n0   = blockIdx.y * 128;

  v8f acc[2][4];
#pragma unroll
  for (int i = 0; i < 2; ++i)
#pragma unroll
    for (int j = 0; j < 4; ++j) acc[i][j] = zero_v8f();

  for (int k0 = 0; k0 < Cc; k0 += 64) {
    // ---- stage A tile (128 x 64) ----
    if constexpr (!A_BF16) {
      const float* A = (const float*)Ap;
      const int r0 = tid >> 3, c4 = (tid & 7) * 4;
#pragma unroll
      for (int rr = 0; rr < 4; ++rr) {
        const int row = rr * 32 + r0;
#pragma unroll
        for (int ch = 0; ch < 2; ++ch) {
          const int col = ch * 32 + c4;
          v4f v = *(const v4f*)&A[(size_t)(m0 + row) * Cc + k0 + col];
#pragma unroll
          for (int i = 0; i < 4; ++i) sA[row][col + i] = (__bf16)v[i];
        }
      }
      if (k0 + 64 < Cc)
        __builtin_prefetch(&A[(size_t)(m0 + r0) * Cc + k0 + 64 + c4], 0, 0);
    } else {
      const __bf16* A = (const __bf16*)Ap;
      const int row = tid >> 1, c = (tid & 1) * 32;
      async_copy_b512(lds_off(&sA[row][c]),
                      &A[(size_t)(m0 + row) * Cc + k0 + c]);
    }
    // ---- stage B tile transposed: sB[k][n] = W[n0+n][k0+k] ----
    {
      const int nn0 = tid >> 3, k4 = (tid & 7) * 4;
#pragma unroll
      for (int p = 0; p < 4; ++p) {
        const int n = p * 32 + nn0;
#pragma unroll
        for (int kh = 0; kh < 64; kh += 32) {
          v4f v = *(const v4f*)&W[(size_t)(n0 + n) * Cc + k0 + kh + k4];
#pragma unroll
          for (int i = 0; i < 4; ++i) sB[kh + k4 + i][n] = (__bf16)v[i];
        }
      }
      if (k0 + 64 < Cc)
        __builtin_prefetch(&W[(size_t)(n0 + nn0) * Cc + k0 + 64 + k4], 0, 0);
    }
    if constexpr (A_BF16) wait_async0();
    __syncthreads();

    // ---- fragments + 16 WMMAs (two K=32 sub-steps) ----
#pragma unroll
    for (int kk = 0; kk < 64; kk += 32) {
      v16bf af[2], bfr[4];
#pragma unroll
      for (int mt = 0; mt < 2; ++mt) {
        const int row = wm * 32 + mt * 16 + ln15;
        af[mt] = ld_frag(&sA[row][kk + half * 8], &sA[row][kk + 16 + half * 8]);
      }
#pragma unroll
      for (int nt = 0; nt < 4; ++nt) {
        const int nb = wn * 64 + nt * 16;
        for (int s = 0; s < 16; ++s) bfr[nt][s] = sB[kk + (s < 8 ? half * 8 + s : 16 + half * 8 + s - 8)][nb + ln15];
      }
#pragma unroll
      for (int mt = 0; mt < 2; ++mt)
#pragma unroll
        for (int nt = 0; nt < 4; ++nt)
          acc[mt][nt] = wmma_bf16(af[mt], bfr[nt], acc[mt][nt]);
    }
    __syncthreads();
  }

  // ---- epilogue ----
#pragma unroll
  for (int mt = 0; mt < 2; ++mt)
#pragma unroll
    for (int nt = 0; nt < 4; ++nt)
#pragma unroll
      for (int j = 0; j < 8; ++j) {
        const int gm = m0 + wm * 32 + mt * 16 + half * 8 + j;
        const int gn = n0 + wn * 64 + nt * 16 + ln15;
        const float v = acc[mt][nt][j];
        if constexpr (SCATTER) {
          __bf16* O = (__bf16*)Op;
          const int b = gm >> 11, t = gm & (Tt - 1);
          const int h = gn >> 6,  d = gn & (Dd - 1);
          O[(((size_t)b * Hh + h) * Tt + t) * Dd + d] = (__bf16)v;
        } else {
          float* O = (float*)Op;
          O[(size_t)gm * Cc + gn] = v;
        }
      }
}

// ---------------------------------------------------------------------------
// Fused causal flash attention over [B,H,T,D] bf16 heads.
// Block = 128 query rows of one (b,h); 8 waves x 16 rows each.
// ---------------------------------------------------------------------------
__global__ __launch_bounds__(256)
void flash_attn_kernel(const __bf16* __restrict__ Qh, const __bf16* __restrict__ Kh,
                       const __bf16* __restrict__ Vh, __bf16* __restrict__ Y) {
  __shared__ __align__(16) __bf16 sKT[64][72];      // [d][kpos] (transposed K chunk)
  __shared__ __align__(16) __bf16 sV[64][72];       // [kpos][d]
  __shared__ __align__(16) __bf16 sP[8][16][72];    // per-wave P staging

  const int tid  = threadIdx.x;
  const int lane = tid & 31;
  const int wv   = tid >> 5;
  const int half = lane >> 4;
  const int ln15 = lane & 15;
  const int q0   = blockIdx.x * 128;
  const int bh   = blockIdx.y;
  const int b    = bh / Hh, h = bh % Hh;

  const __bf16* Qb = Qh + (size_t)bh * Tt * Dd;
  const __bf16* Kb = Kh + (size_t)bh * Tt * Dd;
  const __bf16* Vb = Vh + (size_t)bh * Tt * Dd;

  // Q fragments in registers for the whole kernel (rows q0 + wv*16 + ln15).
  const int qRow = q0 + wv * 16 + ln15;
  v16bf aq[2];
#pragma unroll
  for (int f = 0; f < 2; ++f) {
    const __bf16* p = Qb + (size_t)qRow * Dd + f * 32 + half * 8;
    aq[f] = ld_frag(p, p + 16);
  }

  v8f acc[4];
#pragma unroll
  for (int i = 0; i < 4; ++i) acc[i] = zero_v8f();
  float mrow[8], lrow[8];
#pragma unroll
  for (int j = 0; j < 8; ++j) { mrow[j] = -__builtin_inff(); lrow[j] = 0.0f; }

  const float sc = 0.125f * 1.44269504088896f;  // 1/sqrt(64) * log2(e)

  const int kEnd = q0 + 128;
  for (int kb = 0; kb < kEnd; kb += 64) {
    __syncthreads();  // previous chunk's sKT/sV reads complete before overwrite
    // ---- stage K^T (register transpose) and V (async DMA to LDS) ----
    {
      const int kk = tid >> 2, d0 = (tid & 3) * 16;
      const __bf16* kp = Kb + (size_t)(kb + kk) * Dd + d0;
      v8bf k0v = *(const v8bf*)kp;
      v8bf k1v = *(const v8bf*)(kp + 8);
#pragma unroll
      for (int i = 0; i < 8; ++i) { sKT[d0 + i][kk] = k0v[i]; sKT[d0 + 8 + i][kk] = k1v[i]; }
      const __bf16* vp = Vb + (size_t)(kb + kk) * Dd + d0;
      async_copy_b256(lds_off(&sV[kk][d0]), vp);
      if (kb + 64 < kEnd) {
        __builtin_prefetch(kp + 64 * Dd, 0, 0);
        __builtin_prefetch(vp + 64 * Dd, 0, 0);
      }
    }
    wait_async0();
    __syncthreads();

    // ---- S = Q K^T (16 x 64), contraction over D=64 in two WMMA steps ----
    v8f s[4];
#pragma unroll
    for (int nt = 0; nt < 4; ++nt) {
      v16bf b0, b1;
      for (int s = 0; s < 16; ++s) {
        const int kd = s < 8 ? half * 8 + s : 16 + half * 8 + s - 8;
        b0[s] = sKT[kd][nt * 16 + ln15];
        b1[s] = sKT[32 + kd][nt * 16 + ln15];
      }
      v8f z = zero_v8f();
      z = wmma_bf16(aq[0], b0, z);
      s[nt] = wmma_bf16(aq[1], b1, z);
    }

    // ---- scale, causal mask, chunk row-max ----
    float cmax[8];
#pragma unroll
    for (int j = 0; j < 8; ++j) cmax[j] = -__builtin_inff();
#pragma unroll
    for (int nt = 0; nt < 4; ++nt)
#pragma unroll
      for (int j = 0; j < 8; ++j) {
        float v = s[nt][j] * sc;
        const int kA = kb + nt * 16 + ln15;
        const int qA = q0 + wv * 16 + half * 8 + j;
        if (kA > qA) v = -__builtin_inff();
        s[nt][j] = v;
        cmax[j] = fmaxf(cmax[j], v);
      }
#pragma unroll
    for (int j = 0; j < 8; ++j)
#pragma unroll
      for (int o = 1; o < 16; o <<= 1)
        cmax[j] = fmaxf(cmax[j], __shfl_xor(cmax[j], o, 32));

    float alpha[8], rs[8];
#pragma unroll
    for (int j = 0; j < 8; ++j) {
      const float mn = fmaxf(mrow[j], cmax[j]);
      alpha[j] = exp2f(mrow[j] - mn);
      mrow[j] = mn;
      rs[j] = 0.0f;
    }
    // ---- P = exp2(S - m), stash into per-wave LDS (C-layout -> A-layout) ----
#pragma unroll
    for (int nt = 0; nt < 4; ++nt)
#pragma unroll
      for (int j = 0; j < 8; ++j) {
        const float p = exp2f(s[nt][j] - mrow[j]);
        rs[j] += p;
        sP[wv][half * 8 + j][nt * 16 + ln15] = (__bf16)p;
      }
#pragma unroll
    for (int j = 0; j < 8; ++j) {
#pragma unroll
      for (int o = 1; o < 16; o <<= 1) rs[j] += __shfl_xor(rs[j], o, 32);
      lrow[j] = lrow[j] * alpha[j] + rs[j];
    }
    // ---- rescale O by alpha (row mapping identical to C-layout) ----
#pragma unroll
    for (int ot = 0; ot < 4; ++ot)
#pragma unroll
      for (int j = 0; j < 8; ++j) acc[ot][j] *= alpha[j];

    // Wave-local LDS visibility for sP (cheaper than a block barrier).
    asm volatile("s_wait_dscnt 0" ::: "memory");
    __builtin_amdgcn_wave_barrier();

    // ---- O += P V : contraction over 64 key positions ----
    v16bf pa0 = ld_frag(&sP[wv][ln15][half * 8],      &sP[wv][ln15][16 + half * 8]);
    v16bf pa1 = ld_frag(&sP[wv][ln15][32 + half * 8], &sP[wv][ln15][48 + half * 8]);
#pragma unroll
    for (int ot = 0; ot < 4; ++ot) {
      v16bf vb0, vb1;
      for (int s = 0; s < 16; ++s) {
        const int kd = s < 8 ? half * 8 + s : 16 + half * 8 + s - 8;
        vb0[s] = sV[kd][ot * 16 + ln15];
        vb1[s] = sV[32 + kd][ot * 16 + ln15];
      }
      acc[ot] = wmma_bf16(pa0, vb0, acc[ot]);
      acc[ot] = wmma_bf16(pa1, vb1, acc[ot]);
    }
  }

  // ---- finalize: O / l, write bf16 Y in [B,T,C] layout for the out-proj ----
#pragma unroll
  for (int j = 0; j < 8; ++j) {
    const float inv = 1.0f / lrow[j];
    const int qA = q0 + wv * 16 + half * 8 + j;
#pragma unroll
    for (int ot = 0; ot < 4; ++ot) {
      const int col = h * Dd + ot * 16 + ln15;
      Y[((size_t)b * Tt + qA) * Cc + col] = (__bf16)(acc[ot][j] * inv);
    }
  }
}

// ---------------------------------------------------------------------------
extern "C" void kernel_launch(void* const* d_in, const int* in_sizes, int n_in,
                              void* d_out, int out_size, void* d_ws, size_t ws_size,
                              hipStream_t stream) {
  const float* q  = (const float*)d_in[0];
  const float* k  = (const float*)d_in[1];
  const float* v  = (const float*)d_in[2];
  const float* wq = (const float*)d_in[3];
  const float* wk = (const float*)d_in[4];
  const float* wv = (const float*)d_in[5];
  const float* wo = (const float*)d_in[6];

  const size_t NBHTD = (size_t)Bb * Hh * Tt * Dd;  // 6291456
  __bf16* Qh = (__bf16*)d_ws;
  __bf16* Kh = Qh + NBHTD;
  __bf16* Vh = Kh + NBHTD;
  __bf16* Yb = Vh + NBHTD;  // attention output, [B,T,C] bf16

  const dim3 blk(256);
  const dim3 gProj(Mrows / 128, Cc / 128);  // 64 x 6

  gemm_xwT_kernel<false, true><<<gProj, blk, 0, stream>>>((const void*)q, wq, (void*)Qh);
  gemm_xwT_kernel<false, true><<<gProj, blk, 0, stream>>>((const void*)k, wk, (void*)Kh);
  gemm_xwT_kernel<false, true><<<gProj, blk, 0, stream>>>((const void*)v, wv, (void*)Vh);

  flash_attn_kernel<<<dim3(Tt / 128, Bb * Hh), blk, 0, stream>>>(Qh, Kh, Vh, Yb);

  gemm_xwT_kernel<true, false><<<gProj, blk, 0, stream>>>((const void*)Yb, wo, d_out);
}
